// Attention_Conv_65352222376540
// MI455X (gfx1250) — matrix operand order repaired and checked by reading, not compiled, not run on hardware
//
#include <hip/hip_runtime.h>
#include <hip/hip_bf16.h>

// ---------------------------------------------------------------------------
// Fused BN+ReLU + QKV-projection + attention (f16 WMMA) + depthwise-conv
// residual path for the reference model on gfx1250 (MI455X).
//
// Shapes: b=128, N=256, DIM=256, HEADS=8 (4 attention + 4 conv), HEAD_DIM=64.
// ---------------------------------------------------------------------------

typedef _Float16 h4  __attribute__((ext_vector_type(4)));
typedef _Float16 h8  __attribute__((ext_vector_type(8)));
typedef _Float16 v16h __attribute__((ext_vector_type(16)));
typedef float    v8f  __attribute__((ext_vector_type(8)));

#define KSTR 72    // K buffer row stride in halfs (144B = 36 DW, conflict-free)
#define VSTR 264   // V^T row stride in halfs (528B = 132 DW, conflict-free)
#define ATT_SCALE 0.125f
#define BN_EPS 1e-5f

static __device__ __forceinline__ v8f wmma_f16(v16h a, v16h b, v8f c) {
  // D = A(16x32 f16) x B(32x16 f16) + C(16x16 f32)
  return __builtin_amdgcn_wmma_f32_16x16x32_f16(false, a, false, b,
                                                (short)0, c, false, false);
}
static __device__ __forceinline__ v16h cat8(h8 lo, h8 hi) {
  return __builtin_shufflevector(lo, hi, 0,1,2,3,4,5,6,7,8,9,10,11,12,13,14,15);
}
static __device__ __forceinline__ h8 shflx16_h8(h8 v) {
  union { h8 h; int i[4]; } u; u.h = v;
#pragma unroll
  for (int k = 0; k < 4; ++k) u.i[k] = __shfl_xor(u.i[k], 16, 32);
  return u.h;
}

// ---------------------------------------------------------------------------
// Kernel 1: attention half.  grid = (4 heads, B).  256 threads = 8 waves.
// Wave w handles query M-tiles {w, w+8} (16 rows each).
// Computes scores^T = K * Q^T so that:
//   * K A-fragments  : contiguous ds_load_b128 pairs from row-major K LDS
//   * Q^T B-fragments: contiguous 16-float global loads (BN fused, no LDS)
//   * softmax over keys is in-lane + one shfl_xor(16)
//   * prob fragments re-layout for GEMM2 via one shfl_xor(16) per tile pair
// GEMM2 computes out^T = V^T * attn^T from row-major V^T LDS.
// ---------------------------------------------------------------------------
__global__ __launch_bounds__(256)
void attn_half_kernel(const float* __restrict__ x,
                      const float* __restrict__ bn_g,
                      const float* __restrict__ bn_b,
                      const float* __restrict__ bn_m,
                      const float* __restrict__ bn_v,
                      const float* __restrict__ w_qkv,
                      float* __restrict__ out)
{
  const int hh  = blockIdx.x;        // attention head 0..3
  const int b   = blockIdx.y;
  const int tid = threadIdx.x;
  const int wave = tid >> 5;
  const int lane = tid & 31;
  const int l2   = lane >> 4;        // half-wave index
  const int lm   = lane & 15;
  const int S    = hh * 64;          // feature slice for this head
  const size_t xb0 = (size_t)b * 65536;

  __shared__ _Float16 kb[256 * KSTR];   // K  row-major (n, dd)   36864 B
  __shared__ _Float16 vt[64 * VSTR];    // V^T row-major (dd, n)  33792 B

  // ---- fill K and V^T (one row of x per thread, BN+ReLU fused) ----
  {
    const int n = tid;
    const float a  = bn_g[n] * rsqrtf(bn_v[n] + BN_EPS);
    const float bb = bn_b[n] - bn_m[n] * a;
    const float wk = w_qkv[n * 6 + 2];
    const float wv = w_qkv[n * 6 + 4];
    const float4* xr = (const float4*)(x + xb0 + (size_t)n * 256 + S);
#pragma unroll
    for (int i = 0; i < 16; ++i) {
      float4 p = xr[i];
      float e0 = fmaxf(fmaf(p.x, a, bb), 0.f);
      float e1 = fmaxf(fmaf(p.y, a, bb), 0.f);
      float e2 = fmaxf(fmaf(p.z, a, bb), 0.f);
      float e3 = fmaxf(fmaf(p.w, a, bb), 0.f);
      h4 kh = { (_Float16)(e0 * wk), (_Float16)(e1 * wk),
                (_Float16)(e2 * wk), (_Float16)(e3 * wk) };
      *(h4*)&kb[n * KSTR + 4 * i] = kh;
      vt[(4 * i + 0) * VSTR + n] = (_Float16)(e0 * wv);
      vt[(4 * i + 1) * VSTR + n] = (_Float16)(e1 * wv);
      vt[(4 * i + 2) * VSTR + n] = (_Float16)(e2 * wv);
      vt[(4 * i + 3) * VSTR + n] = (_Float16)(e3 * wv);
    }
  }
  __syncthreads();

#pragma unroll
  for (int it = 0; it < 2; ++it) {
    const int mt = wave + 8 * it;        // query M-tile
    const int mg = mt * 16 + lm;         // this lane's query row (B-frag col)

    // ---- Q^T B-fragments straight from global x (BN+ReLU+wq fused) ----
    const float aq  = bn_g[mg] * rsqrtf(bn_v[mg] + BN_EPS);
    const float bq  = bn_b[mg] - bn_m[mg] * aq;
    const float wq  = w_qkv[mg * 6 + 0];
    v16h qf[2];
#pragma unroll
    for (int ks = 0; ks < 2; ++ks) {
      const float4* qp = (const float4*)(x + xb0 + (size_t)mg * 256 + S
                                         + ks * 32 + 16 * l2);
      v16h qv;
#pragma unroll
      for (int i2 = 0; i2 < 4; ++i2) {
        float4 p = qp[i2];
        qv[4 * i2 + 0] = (_Float16)(fmaxf(fmaf(p.x, aq, bq), 0.f) * wq);
        qv[4 * i2 + 1] = (_Float16)(fmaxf(fmaf(p.y, aq, bq), 0.f) * wq);
        qv[4 * i2 + 2] = (_Float16)(fmaxf(fmaf(p.z, aq, bq), 0.f) * wq);
        qv[4 * i2 + 3] = (_Float16)(fmaxf(fmaf(p.w, aq, bq), 0.f) * wq);
      }
      qf[ks] = qv;
    }

    // ---- GEMM1: scores^T tiles (key-tile nt rows x query-col lm) ----
    v8f sc[16];
#pragma unroll
    for (int nt = 0; nt < 16; ++nt) {
      v8f c = {0.f,0.f,0.f,0.f,0.f,0.f,0.f,0.f};
#pragma unroll
      for (int ks = 0; ks < 2; ++ks) {
        const _Float16* kr = &kb[(nt * 16 + lm) * KSTR + ks * 32 + 16 * l2];
        h8 lo = *(const h8*)kr;
        h8 hi = *(const h8*)(kr + 8);
        c = wmma_f16(cat8(lo, hi), qf[ks], c);
      }
      sc[nt] = c;
    }

    // ---- softmax over keys (rows of scores^T) for this lane's column m ----
    float mx = -3.4e38f;
#pragma unroll
    for (int nt = 0; nt < 16; ++nt)
#pragma unroll
      for (int i = 0; i < 8; ++i)
        mx = fmaxf(mx, sc[nt][i] * ATT_SCALE);
    mx = fmaxf(mx, __shfl_xor(mx, 16, 32));

    float sum = 0.f;
    h8 ph[16];
#pragma unroll
    for (int nt = 0; nt < 16; ++nt) {
#pragma unroll
      for (int i = 0; i < 8; ++i) {
        float p = __expf(fmaf(sc[nt][i], ATT_SCALE, -mx));
        sum += p;
        ph[nt][i] = (_Float16)p;
      }
    }
    sum += __shfl_xor(sum, 16, 32);
    const float s5 = 0.5f / sum;   // 1/rowsum folded with the pair-average

    // ---- GEMM2: out^T = V^T x attn^T ----
    v8f oc[4];
#pragma unroll
    for (int a4 = 0; a4 < 4; ++a4)
      oc[a4] = (v8f){0.f,0.f,0.f,0.f,0.f,0.f,0.f,0.f};

#pragma unroll
    for (int jt = 0; jt < 8; ++jt) {
      // rebuild attn^T B-fragment from the C-layout prob tiles:
      // lane needs all 16 rows of tile (2*jt + l2); it owns rows 8*l2..+7 of
      // every tile, partner lane (xor 16) supplies the other 8.
      h8 keep = l2 ? ph[2 * jt + 1] : ph[2 * jt];
      h8 send = l2 ? ph[2 * jt]     : ph[2 * jt + 1];
      h8 recv = shflx16_h8(send);
      h8 lo8  = l2 ? recv : keep;   // rows 0..7 of needed tile
      h8 hi8  = l2 ? keep : recv;   // rows 8..15
      v16h bf = cat8(lo8, hi8);
#pragma unroll
      for (int a4 = 0; a4 < 4; ++a4) {
        const _Float16* vr = &vt[(a4 * 16 + lm) * VSTR + jt * 32 + 16 * l2];
        h8 alo = *(const h8*)vr;
        h8 ahi = *(const h8*)(vr + 8);
        oc[a4] = wmma_f16(cat8(alo, ahi), bf, oc[a4]);
      }
    }

    // ---- normalize + pair-average + store (float4, coalesced) ----
    float* ob = out + xb0 + (size_t)mg * 256 + hh * 32;
#pragma unroll
    for (int a4 = 0; a4 < 4; ++a4) {
      float4 o;
      o.x = (oc[a4][0] + oc[a4][1]) * s5;
      o.y = (oc[a4][2] + oc[a4][3]) * s5;
      o.z = (oc[a4][4] + oc[a4][5]) * s5;
      o.w = (oc[a4][6] + oc[a4][7]) * s5;
      *(float4*)(ob + a4 * 8 + 4 * l2) = o;
    }
  }
}

// ---------------------------------------------------------------------------
// Kernel 2: conv-residual half (output dims 128..255).
// grid = (64, B), 256 threads. WG (b,q) needs only xb rows 4q..4q+3, all cols.
// c2 = hh2*64 + q ; s2 = 64*r + dd ; xb source = (row 4q+r, col hh2*64+dd).
// q-conv: 1x3 along dd%16 ; k-conv: 3x1 along u = 4r + dd/16 (crosses rows).
// ---------------------------------------------------------------------------
__global__ __launch_bounds__(256)
void conv_half_kernel(const float* __restrict__ x,
                      const float* __restrict__ bn_g,
                      const float* __restrict__ bn_b,
                      const float* __restrict__ bn_m,
                      const float* __restrict__ bn_v,
                      const float* __restrict__ w_qkv,
                      const float* __restrict__ w_qs,
                      const float* __restrict__ w_ks,
                      float* __restrict__ out)
{
  const int qq = blockIdx.x;
  const int b  = blockIdx.y;
  const int t  = threadIdx.x;
  const size_t xb0 = (size_t)b * 65536;

  __shared__ float xbt[4][256];

#pragma unroll
  for (int i = 0; i < 4; ++i) {
    const int n = 4 * qq + i;
    const float a  = bn_g[n] * rsqrtf(bn_v[n] + BN_EPS);
    const float bb = bn_b[n] - bn_m[n] * a;
    const float v  = x[xb0 + (size_t)n * 256 + t];
    xbt[i][t] = fmaxf(fmaf(v, a, bb), 0.f);
  }
  __syncthreads();

  const int hh2 = t >> 6;            // conv head 0..3
  const int g   = t & 63;            // covers s2 = 4g .. 4g+3
  const int c2  = hh2 * 64 + qq;     // output row
  const int S   = hh2 * 64;

  const float qs0 = w_qs[c2 * 3 + 0], qs1 = w_qs[c2 * 3 + 1], qs2 = w_qs[c2 * 3 + 2];
  const float ks0 = w_ks[c2 * 3 + 0], ks1 = w_ks[c2 * 3 + 1], ks2 = w_ks[c2 * 3 + 2];

  const int s0  = 4 * g;
  const int r   = s0 >> 6;           // constant over the 4 outputs
  const int dd0 = s0 & 63;
  const int u   = 4 * r + (dd0 >> 4);
  const int nb  = 4 * qq;

  const float w1r = w_qkv[(nb + r) * 6 + 1];
  const float w3r = w_qkv[(nb + r) * 6 + 3];
  const float w5r = w_qkv[(nb + r) * 6 + 5];

  float res[4];
#pragma unroll
  for (int j = 0; j < 4; ++j) {
    const int dd = dd0 + j;
    const int w2 = dd & 15;
    const float xv = xbt[r][S + dd];
    // v2 term
    float acc = xv * w5r;
    // q-conv (1x3 along w2, same row)
    float qa = xv * qs1;
    if (w2 > 0)  qa += xbt[r][S + dd - 1] * qs0;
    if (w2 < 15) qa += xbt[r][S + dd + 1] * qs2;
    acc = fmaf(qa, w1r, acc);
    // k-conv (3x1 along u; neighbors may live in other rows of the 4-block)
    float ka = xv * ks1 * w3r;
    if (u > 0) {
      const int up = u - 1, rp = up >> 2;
      const int ddp = ((up & 3) << 4) + w2;
      ka = fmaf(xbt[rp][S + ddp] * w_qkv[(nb + rp) * 6 + 3], ks0, ka);
    }
    if (u < 15) {
      const int un = u + 1, rn = un >> 2;
      const int ddn = ((un & 3) << 4) + w2;
      ka = fmaf(xbt[rn][S + ddn] * w_qkv[(nb + rn) * 6 + 3], ks2, ka);
    }
    res[j] = acc + ka;
  }

  float2 o;
  o.x = (res[0] + res[1]) * 0.5f;
  o.y = (res[2] + res[3]) * 0.5f;
  *(float2*)(out + xb0 + (size_t)c2 * 256 + 128 + 2 * g) = o;
}

// ---------------------------------------------------------------------------
extern "C" void kernel_launch(void* const* d_in, const int* in_sizes, int n_in,
                              void* d_out, int out_size, void* d_ws, size_t ws_size,
                              hipStream_t stream)
{
  (void)n_in; (void)out_size; (void)d_ws; (void)ws_size;
  const float* x     = (const float*)d_in[0];
  const float* bn_g  = (const float*)d_in[1];
  const float* bn_b  = (const float*)d_in[2];
  const float* bn_m  = (const float*)d_in[3];
  const float* bn_v  = (const float*)d_in[4];
  const float* w_qkv = (const float*)d_in[5];
  const float* w_qs  = (const float*)d_in[6];
  const float* w_ks  = (const float*)d_in[7];
  float* out = (float*)d_out;

  const int B = in_sizes[0] / 65536;   // b = 128 for the reference shapes

  dim3 g1(4, B), g2(64, B), blk(256);
  attn_half_kernel<<<g1, blk, 0, stream>>>(x, bn_g, bn_b, bn_m, bn_v, w_qkv, out);
  conv_half_kernel<<<g2, blk, 0, stream>>>(x, bn_g, bn_b, bn_m, bn_v,
                                           w_qkv, w_qs, w_ks, out);
}
